// FlashGQA_83124797046958
// MI455X (gfx1250) — matrix operand order repaired and checked by reading, not compiled, not run on hardware
//
#include <hip/hip_runtime.h>
#include <hip/hip_bf16.h>
#include <cstdint>
#include <cstddef>

// ---------------------------------------------------------------------------
// Problem constants (match reference)
// ---------------------------------------------------------------------------
#define Bb 2
#define Ss 2048
#define Dd 2048
#define NH 32
#define KVH 8
#define HD 64
#define GRP 4

typedef __bf16 bf16_t;
typedef bf16_t v16bf __attribute__((ext_vector_type(16)));
typedef bf16_t v8bf  __attribute__((ext_vector_type(8)));
typedef bf16_t v4bf  __attribute__((ext_vector_type(4)));
typedef float  v8f   __attribute__((ext_vector_type(8)));
typedef int    v4i   __attribute__((ext_vector_type(4)));

#define SHUF16(lo, hi) __builtin_shufflevector((lo), (hi), 0,1,2,3,4,5,6,7,8,9,10,11,12,13,14,15)

__device__ __forceinline__ v8f wmma_bf16(v16bf a, v16bf b, v8f c) {
  return __builtin_amdgcn_wmma_f32_16x16x32_bf16(false, a, false, b, (short)0, c, false, false);
}

// ---------------------------------------------------------------------------
// CDNA5 async global->LDS copy (ASYNCcnt-tracked DMA, no VGPR round-trip).
// Probe result: builtin takes (int4 AS1*, int4 AS3*, imm offset, imm cpol).
// Guarded so a toolchain without the builtin still compiles via the sync path.
// ---------------------------------------------------------------------------
#if defined(__has_builtin)
#  if __has_builtin(__builtin_amdgcn_global_load_async_to_lds_b128) && \
      __has_builtin(__builtin_amdgcn_s_wait_asynccnt)
#    define HAVE_ASYNC_LDS 1
#  endif
#endif
#ifndef HAVE_ASYNC_LDS
#  define HAVE_ASYNC_LDS 0
#endif

typedef __attribute__((address_space(1))) v4i* gptr4;
typedef __attribute__((address_space(3))) v4i* lptr4;

__device__ __forceinline__ void cp_async16(const bf16_t* g, bf16_t* l) {
#if HAVE_ASYNC_LDS
  __builtin_amdgcn_global_load_async_to_lds_b128((gptr4)(g), (lptr4)(l), 0, 0);
#else
  *(v8bf*)l = *(const v8bf*)g;
#endif
}

__device__ __forceinline__ void async_wait0() {
#if HAVE_ASYNC_LDS
  __builtin_amdgcn_s_wait_asynccnt(0);
#endif
}

// LDS pad strides (halves). Rows stay 16B-aligned for ds_load_b128 and the 16
// rows of one fragment read land in 16 distinct 4-bank groups.
#define LDSW_G 40   // GEMM: 32 cols + 8 pad  (80B rows)
#define LDSW_F 72   // Flash: 64 cols + 8 pad (144B rows)

// ---------------------------------------------------------------------------
// Kernel 0: fp32 -> bf16 conversion (vectorized x4)
// ---------------------------------------------------------------------------
__global__ __launch_bounds__(256) void cvt_f32_bf16(const float* __restrict__ in,
                                                    bf16_t* __restrict__ out, int n4) {
  int i = blockIdx.x * 256 + threadIdx.x;
  if (i < n4) {
    float4 f = ((const float4*)in)[i];
    v4bf o;
    o[0] = (bf16_t)f.x; o[1] = (bf16_t)f.y; o[2] = (bf16_t)f.z; o[3] = (bf16_t)f.w;
    ((v4bf*)out)[i] = o;
  }
}

// ---------------------------------------------------------------------------
// Shared GEMM mainloop: C(128x128) = A(128xK) * B(128xK)^T, bf16, f32 accum.
// 256 threads = 8 waves, wave grid 4(M) x 2(N); each wave: 32x64 = 2x4 tiles.
// BK=32, double-buffered async DMA into LDS.
// As/Bs: 2 buffers of 128*LDSW_G halves each.
// ---------------------------------------------------------------------------
__device__ __forceinline__ void gemm_mainloop(const bf16_t* __restrict__ A,
                                              const bf16_t* __restrict__ Bm,
                                              int Kdim, int lda, int ldb,
                                              int m0, int n0,
                                              bf16_t* As, bf16_t* Bs,
                                              v8f acc[2][4]) {
  const int tid  = threadIdx.x;
  const int wave = tid >> 5, lane = tid & 31;
  const int wm = wave >> 1, wn = wave & 1;
  const int hh = lane >> 4, l15 = lane & 15;

#pragma unroll
  for (int im = 0; im < 2; ++im)
#pragma unroll
    for (int in_ = 0; in_ < 4; ++in_)
#pragma unroll
      for (int v = 0; v < 8; ++v) acc[im][in_][v] = 0.0f;

  // Issue async DMA of one 128x32 A-slab + B-slab into buffer `buf`.
  auto issue = [&](int k0, int buf) {
    bf16_t* as = As + buf * (128 * LDSW_G);
    bf16_t* bs = Bs + buf * (128 * LDSW_G);
#pragma unroll
    for (int i = 0; i < 2; ++i) {
      int idx = tid + i * 256;           // 0..511
      int r = idx >> 2, c = (idx & 3) * 8;
      cp_async16(&A[(size_t)(m0 + r) * lda + k0 + c], &as[r * LDSW_G + c]);
      cp_async16(&Bm[(size_t)(n0 + r) * ldb + k0 + c], &bs[r * LDSW_G + c]);
    }
  };

  issue(0, 0);
  for (int k0 = 0; k0 < Kdim; k0 += 32) {
    const int cur = (k0 >> 5) & 1;
    async_wait0();          // my DMA into `cur` done
    __syncthreads();        // everyone's DMA done; prev reads of cur^1 done
    if (k0 + 32 < Kdim) issue(k0 + 32, cur ^ 1);

    const bf16_t* as = As + cur * (128 * LDSW_G);
    const bf16_t* bs = Bs + cur * (128 * LDSW_G);

    v16bf af[2];
#pragma unroll
    for (int im = 0; im < 2; ++im) {
      int r = wm * 32 + im * 16 + l15;
      v8bf lo = *(const v8bf*)&as[r * LDSW_G + hh * 8];
      v8bf hi = *(const v8bf*)&as[r * LDSW_G + hh * 8 + 16];
      af[im] = SHUF16(lo, hi);
    }
    v16bf bfv[4];
#pragma unroll
    for (int in_ = 0; in_ < 4; ++in_) {
      int r = wn * 64 + in_ * 16 + l15;
      v8bf lo = *(const v8bf*)&bs[r * LDSW_G + hh * 8];
      v8bf hi = *(const v8bf*)&bs[r * LDSW_G + hh * 8 + 16];
      bfv[in_] = SHUF16(lo, hi);
    }
#pragma unroll
    for (int im = 0; im < 2; ++im)
#pragma unroll
      for (int in_ = 0; in_ < 4; ++in_)
        acc[im][in_] = wmma_bf16(af[im], bfv[in_], acc[im][in_]);
  }
}

// ---------------------------------------------------------------------------
// Kernel 1: fused QKV projection + RoPE.
// N space: [0,2048)=Q, [2048,2560)=K, [2560,3072)=V. Writes bf16:
//   Qh[b][h][s][d], Kh[b][kvh][s][d], Vt[b][kvh][d][s] (V transposed).
// ---------------------------------------------------------------------------
__global__ __launch_bounds__(256) void qkv_rope_kernel(const bf16_t* __restrict__ xb,
                                                       const bf16_t* __restrict__ wq,
                                                       const bf16_t* __restrict__ wk,
                                                       const bf16_t* __restrict__ wv,
                                                       bf16_t* __restrict__ qh,
                                                       bf16_t* __restrict__ kh,
                                                       bf16_t* __restrict__ vt) {
  __shared__ __align__(16) bf16_t As[2 * 128 * LDSW_G];
  __shared__ __align__(16) bf16_t Bs[2 * 128 * LDSW_G];

  const int n0g = blockIdx.x * 128;
  const int m0  = blockIdx.y * 128;
  int seg, nb;
  const bf16_t* W;
  if (n0g < 2048)      { seg = 0; W = wq; nb = n0g; }
  else if (n0g < 2560) { seg = 1; W = wk; nb = n0g - 2048; }
  else                 { seg = 2; W = wv; nb = n0g - 2560; }

  v8f acc[2][4];
  gemm_mainloop(xb, W, Dd, Dd, Dd, m0, nb, As, Bs, acc);

  const int lane = threadIdx.x & 31;
  const int wave = threadIdx.x >> 5;
  const int wm = wave >> 1, wn = wave & 1;
  const int hh = lane >> 4, l15 = lane & 15;

#pragma unroll
  for (int im = 0; im < 2; ++im)
#pragma unroll
    for (int in_ = 0; in_ < 4; ++in_)
#pragma unroll
      for (int v = 0; v < 8; ++v) {
        int m = m0 + wm * 32 + im * 16 + hh * 8 + v;   // global row (b*S + s)
        int n = nb + wn * 64 + in_ * 16 + l15;          // col within segment
        float val = acc[im][in_][v];
        int bidx = m >> 11, srow = m & (Ss - 1);
        int d = n & 63;
        // RoPE pairing across even/odd columns (adjacent lanes).
        float partner = __shfl_xor(val, 1, 32);
        int   p   = d >> 1;
        float inv = __powf(10000.0f, -(float)p * (1.0f / 32.0f));
        float sn, cs;
        __sincosf((float)srow * inv, &sn, &cs);
        float rope = (d & 1) ? (partner * sn + val * cs)
                             : (val * cs - partner * sn);
        if (seg == 0) {
          int h = n >> 6;
          qh[((size_t)(bidx * NH + h) * Ss + srow) * HD + d] = (bf16_t)rope;
        } else if (seg == 1) {
          int kvh = n >> 6;
          kh[((size_t)(bidx * KVH + kvh) * Ss + srow) * HD + d] = (bf16_t)rope;
        } else {
          int kvh = n >> 6;
          vt[((size_t)(bidx * KVH + kvh) * HD + d) * Ss + srow] = (bf16_t)val;
        }
      }
}

// ---------------------------------------------------------------------------
// Kernel 2: flash attention (causal, GQA). Block = (qt, head, b), 128 threads
// = 4 waves; wave w owns q-rows [64*qt + 16w, +16). Double-buffered async K/V
// tiles in LDS, Q fragments in registers, online softmax in f32.
// ---------------------------------------------------------------------------
__global__ __launch_bounds__(128) void flash_gqa_kernel(const bf16_t* __restrict__ Qh,
                                                        const bf16_t* __restrict__ Kh,
                                                        const bf16_t* __restrict__ Vt,
                                                        bf16_t* __restrict__ Ob) {
  __shared__ __align__(16) bf16_t Ks[2 * 64 * LDSW_F];
  __shared__ __align__(16) bf16_t Vs[2 * 64 * LDSW_F];
  __shared__ __align__(16) bf16_t Ps[64 * LDSW_F];

  const int qt = blockIdx.x, h = blockIdx.y, b = blockIdx.z;
  const int kvh = h >> 2;
  const int tid = threadIdx.x, wave = tid >> 5, lane = tid & 31;
  const int hh = lane >> 4, l15 = lane & 15;
  const int q0 = qt * 64;

  // Q fragments (16 rows x 64 contraction) -> 2 A-frags, kept in registers.
  const size_t qbase = ((size_t)(b * NH + h) * Ss + (q0 + wave * 16 + l15)) * HD;
  v16bf qf[2];
#pragma unroll
  for (int f = 0; f < 2; ++f) {
    v8bf lo = *(const v8bf*)&Qh[qbase + f * 32 + hh * 8];
    v8bf hi = *(const v8bf*)&Qh[qbase + f * 32 + hh * 8 + 16];
    qf[f] = SHUF16(lo, hi);
  }

  v8f o_acc[4];
  float mrow[8], lrow[8];
#pragma unroll
  for (int j = 0; j < 4; ++j)
#pragma unroll
    for (int v = 0; v < 8; ++v) o_acc[j][v] = 0.0f;
#pragma unroll
  for (int v = 0; v < 8; ++v) { mrow[v] = -3.0e38f; lrow[v] = 0.0f; }

  const size_t kvbase = (size_t)(b * KVH + kvh);

  // Async DMA of K (64s x 64d) and Vt (64d x 64s) tiles for step kt.
  auto issueKV = [&](int kt, int buf) {
    bf16_t* ks = Ks + buf * (64 * LDSW_F);
    bf16_t* vs = Vs + buf * (64 * LDSW_F);
#pragma unroll
    for (int i = 0; i < 4; ++i) {
      int idx = tid + i * 128;            // 0..511
      int r = idx >> 3, c = (idx & 7) * 8;
      cp_async16(&Kh[(kvbase * Ss + (size_t)(kt * 64 + r)) * HD + c],
                 &ks[r * LDSW_F + c]);
      cp_async16(&Vt[(kvbase * HD + (size_t)r) * Ss + kt * 64 + c],
                 &vs[r * LDSW_F + c]);
    }
  };

  issueKV(0, 0);
  for (int kt = 0; kt <= qt; ++kt) {
    const int cur = kt & 1;
    async_wait0();
    __syncthreads();
    if (kt < qt) issueKV(kt + 1, cur ^ 1);

    const bf16_t* ks = Ks + cur * (64 * LDSW_F);
    const bf16_t* vs = Vs + cur * (64 * LDSW_F);

    // S = Q * K^T  (contraction over d=64, two 32-chunks).
    v8f s_acc[4];
#pragma unroll
    for (int j = 0; j < 4; ++j) {
      v8f a;
#pragma unroll
      for (int v = 0; v < 8; ++v) a[v] = 0.0f;
#pragma unroll
      for (int f = 0; f < 2; ++f) {
        int r = j * 16 + l15;
        v8bf lo = *(const v8bf*)&ks[r * LDSW_F + f * 32 + hh * 8];
        v8bf hi = *(const v8bf*)&ks[r * LDSW_F + f * 32 + hh * 8 + 16];
        a = wmma_bf16(qf[f], SHUF16(lo, hi), a);
      }
      s_acc[j] = a;
    }

    // Scale + causal mask (only diagonal tile needs masking).
#pragma unroll
    for (int j = 0; j < 4; ++j)
#pragma unroll
      for (int v = 0; v < 8; ++v) {
        float sv = s_acc[j][v] * 0.125f;   // 1/sqrt(64)
        if (kt == qt) {
          int kg = (kt << 6) + (j << 4) + l15;
          int qg = q0 + (wave << 4) + (hh << 3) + v;
          if (kg > qg) sv = -1.0e30f;
        }
        s_acc[j][v] = sv;
      }

    // Online softmax: per-row stats via 16-lane xor-shuffle reductions.
#pragma unroll
    for (int v = 0; v < 8; ++v) {
      float t = fmaxf(fmaxf(s_acc[0][v], s_acc[1][v]),
                      fmaxf(s_acc[2][v], s_acc[3][v]));
#pragma unroll
      for (int msk = 8; msk >= 1; msk >>= 1) t = fmaxf(t, __shfl_xor(t, msk, 32));
      float mnew  = fmaxf(mrow[v], t);
      float alpha = __expf(mrow[v] - mnew);
      mrow[v] = mnew;
      float rsum = 0.0f;
#pragma unroll
      for (int j = 0; j < 4; ++j) {
        float p = __expf(s_acc[j][v] - mnew);
        s_acc[j][v] = p;
        rsum += p;
      }
#pragma unroll
      for (int msk = 8; msk >= 1; msk >>= 1) rsum += __shfl_xor(rsum, msk, 32);
      lrow[v] = lrow[v] * alpha + rsum;
#pragma unroll
      for (int j = 0; j < 4; ++j) o_acc[j][v] *= alpha;
    }

    // P -> LDS (bf16) for A-fragment re-layout (wave-private rows; LDS ops
    // from the same wave are processed in order).
#pragma unroll
    for (int j = 0; j < 4; ++j)
#pragma unroll
      for (int v = 0; v < 8; ++v) {
        int r = wave * 16 + hh * 8 + v;
        int c = j * 16 + l15;
        Ps[r * LDSW_F + c] = (bf16_t)s_acc[j][v];
      }

    // O += P * V  (contraction over k=64, two 32-chunks).
#pragma unroll
    for (int f = 0; f < 2; ++f) {
      int r = wave * 16 + l15;
      v8bf plo = *(const v8bf*)&Ps[r * LDSW_F + f * 32 + hh * 8];
      v8bf phi = *(const v8bf*)&Ps[r * LDSW_F + f * 32 + hh * 8 + 16];
      v16bf pa = SHUF16(plo, phi);
#pragma unroll
      for (int j = 0; j < 4; ++j) {
        int rv = j * 16 + l15;
        v8bf blo = *(const v8bf*)&vs[rv * LDSW_F + f * 32 + hh * 8];
        v8bf bhi = *(const v8bf*)&vs[rv * LDSW_F + f * 32 + hh * 8 + 16];
        o_acc[j] = wmma_bf16(pa, SHUF16(blo, bhi), o_acc[j]);
      }
    }
  }

  // Normalize and store O (bf16) in [b][s][h*64+d] layout for the O-proj GEMM.
#pragma unroll
  for (int j = 0; j < 4; ++j)
#pragma unroll
    for (int v = 0; v < 8; ++v) {
      int srow = q0 + wave * 16 + hh * 8 + v;
      int d    = j * 16 + l15;
      float o  = o_acc[j][v] / lrow[v];
      Ob[((size_t)b * Ss + srow) * Dd + h * HD + d] = (bf16_t)o;
    }
}

// ---------------------------------------------------------------------------
// Kernel 3: output projection, fp32 result to d_out.
// ---------------------------------------------------------------------------
__global__ __launch_bounds__(256) void oproj_kernel(const bf16_t* __restrict__ Ob,
                                                    const bf16_t* __restrict__ wo,
                                                    float* __restrict__ out) {
  __shared__ __align__(16) bf16_t As[2 * 128 * LDSW_G];
  __shared__ __align__(16) bf16_t Bs[2 * 128 * LDSW_G];

  const int n0 = blockIdx.x * 128;
  const int m0 = blockIdx.y * 128;

  v8f acc[2][4];
  gemm_mainloop(Ob, wo, Dd, Dd, Dd, m0, n0, As, Bs, acc);

  const int lane = threadIdx.x & 31;
  const int wave = threadIdx.x >> 5;
  const int wm = wave >> 1, wn = wave & 1;
  const int hh = lane >> 4, l15 = lane & 15;

#pragma unroll
  for (int im = 0; im < 2; ++im)
#pragma unroll
    for (int in_ = 0; in_ < 4; ++in_)
#pragma unroll
      for (int v = 0; v < 8; ++v) {
        int m = m0 + wm * 32 + im * 16 + hh * 8 + v;
        int n = n0 + wn * 64 + in_ * 16 + l15;
        out[(size_t)m * Dd + n] = acc[im][in_][v];
      }
}

// ---------------------------------------------------------------------------
// Host-side launcher
// ---------------------------------------------------------------------------
extern "C" void kernel_launch(void* const* d_in, const int* in_sizes, int n_in,
                              void* d_out, int out_size, void* d_ws, size_t ws_size,
                              hipStream_t stream) {
  (void)in_sizes; (void)n_in; (void)out_size; (void)ws_size;

  const float* x  = (const float*)d_in[0];
  const float* Wq = (const float*)d_in[1];
  const float* Wk = (const float*)d_in[2];
  const float* Wv = (const float*)d_in[3];
  const float* Wo = (const float*)d_in[4];

  char* ws = (char*)d_ws;
  size_t off = 0;
  auto alloc = [&](size_t bytes) -> char* {
    char* p = ws + off;
    off += (bytes + 255) & ~(size_t)255;
    return p;
  };

  const size_t N_X  = (size_t)Bb * Ss * Dd;        // 8.4M
  const size_t N_WQ = (size_t)Dd * Dd;             // 4.2M
  const size_t N_WK = (size_t)(KVH * HD) * Dd;     // 1.05M
  const size_t N_QH = (size_t)Bb * NH * Ss * HD;   // 8.4M
  const size_t N_KH = (size_t)Bb * KVH * Ss * HD;  // 2.1M

  bf16_t* xb  = (bf16_t*)alloc(N_X  * 2);
  bf16_t* wqb = (bf16_t*)alloc(N_WQ * 2);
  bf16_t* wkb = (bf16_t*)alloc(N_WK * 2);
  bf16_t* wvb = (bf16_t*)alloc(N_WK * 2);
  bf16_t* wob = (bf16_t*)alloc(N_WQ * 2);
  bf16_t* qh  = (bf16_t*)alloc(N_QH * 2);
  bf16_t* kh  = (bf16_t*)alloc(N_KH * 2);
  bf16_t* vt  = (bf16_t*)alloc(N_KH * 2);
  bf16_t* ob  = (bf16_t*)alloc(N_X  * 2);

  auto cvt = [&](const float* src, bf16_t* dst, size_t n) {
    int n4 = (int)(n / 4);
    cvt_f32_bf16<<<(n4 + 255) / 256, 256, 0, stream>>>(src, dst, n4);
  };
  cvt(x,  xb,  N_X);
  cvt(Wq, wqb, N_WQ);
  cvt(Wk, wkb, N_WK);
  cvt(Wv, wvb, N_WK);
  cvt(Wo, wob, N_WQ);

  // QKV + RoPE: M = B*S = 4096 rows (32 tiles), N = 3072 (24 tiles).
  qkv_rope_kernel<<<dim3(24, 32), 256, 0, stream>>>(xb, wqb, wkb, wvb, qh, kh, vt);

  // Flash attention: (S/64, heads, B) blocks of 128 threads.
  flash_gqa_kernel<<<dim3(Ss / 64, NH, Bb), 128, 0, stream>>>(qh, kh, vt, ob);

  // Output projection: M = 4096 (32 tiles), N = 2048 (16 tiles).
  oproj_kernel<<<dim3(16, 32), 256, 0, stream>>>(ob, wob, (float*)d_out);
}
